// PGBF_Surv01_68539088110085
// MI455X (gfx1250) — matrix operand order repaired and checked by reading, not compiled, not run on hardware
//
#include <hip/hip_runtime.h>

typedef __bf16 bf16_t;
typedef __attribute__((ext_vector_type(8)))  __bf16 v8bf;
typedef __attribute__((ext_vector_type(16))) __bf16 v16bf;
typedef __attribute__((ext_vector_type(8)))  float  v8f;

#define NPATCH 8192
#define DDIM   256
#define KTOP   8
#define SCALEF 0.0625f   // 256^-0.5

// ---------------------------------------------------------------------------
// WMMA fragment loaders (wave32, 16x16x32 bf16).
// A-matrix 16x32: lane L<16 holds row L, K={k0..k0+7, k0+16..k0+23};
//                 lane L>=16 holds row L-16, K={k0+8..k0+15, k0+24..k0+31}.
// B-matrix 32x16: lane L<16 holds col L,  K={k0..k0+7, k0+16..k0+23};
//                 lane L>=16 holds col L-16, K={k0+8..k0+15, k0+24..k0+31}.
// Both operands come from row-major [rows, K] storage (A rows / W^T columns).
// ---------------------------------------------------------------------------
__device__ inline v16bf load_a_frag(const bf16_t* __restrict__ A, int ld,
                                    int row0, int k0, int lane) {
  int r  = row0 + (lane & 15);
  int kb = k0 + ((lane >> 4) << 3);
  const bf16_t* p = A + (size_t)r * ld + kb;
  v8bf lo = *reinterpret_cast<const v8bf*>(p);
  v8bf hi = *reinterpret_cast<const v8bf*>(p + 16);
  v16bf f;
#pragma unroll
  for (int i = 0; i < 8; ++i) { f[i] = lo[i]; f[8 + i] = hi[i]; }
  return f;
}

__device__ inline v16bf load_b_frag(const bf16_t* __restrict__ W, int ld,
                                    int col0, int k0, int lane) {
  int n  = col0 + (lane & 15);
  int kb = k0 + ((lane >> 4) << 3);
  const bf16_t* p = W + (size_t)n * ld + kb;
  v8bf lo = *reinterpret_cast<const v8bf*>(p);
  v8bf hi = *reinterpret_cast<const v8bf*>(p + 16);
  v16bf f;
#pragma unroll
  for (int i = 0; i < 8; ++i) { f[i] = lo[i]; f[8 + i] = hi[i]; }
  return f;
}

#define WMMA_BF16(a, b, c) \
  __builtin_amdgcn_wmma_f32_16x16x32_bf16(false, (a), false, (b), (short)0, (c), false, false)

__device__ inline void store_tile(const v8f& acc, int row0, int col0, int N,
                                  const float* __restrict__ bias,
                                  const float* __restrict__ addsrc,
                                  float* __restrict__ outf,
                                  bf16_t* __restrict__ outb, int act, int lane) {
  int col   = col0 + (lane & 15);
  int rbase = row0 + ((lane >> 4) << 3);
  float bval = bias ? bias[col] : 0.0f;
#pragma unroll
  for (int j = 0; j < 8; ++j) {
    int row = rbase + j;
    float v = acc[j] + bval;
    if (act == 1)      v = fmaxf(v, 0.0f);
    else if (act == 2) v = (v >= 0.0f) ? v : 0.01f * v;
    size_t o = (size_t)row * N + col;
    if (addsrc) v += addsrc[o];
    if (outf) outf[o] = v;
    if (outb) outb[o] = (bf16_t)v;
  }
}

// ---------------------------------------------------------------------------
// Register-blocked GEMM: out = act(A[M,K] @ W[N,K]^T + bias) (+ addsrc)
// act: 0 = none, 1 = relu, 2 = leaky_relu(0.01)
// One wave computes a 32x32 tile (2x2 accumulators): 4 fragment loads feed
// 4 WMMAs per k-step, doubling FLOP/byte vs wave-per-16x16.
// ---------------------------------------------------------------------------
__global__ __launch_bounds__(256) void gemm_bf16_kernel(
    const bf16_t* __restrict__ A, const bf16_t* __restrict__ W,
    const float* __restrict__ bias, const float* __restrict__ addsrc,
    float* __restrict__ outf, bf16_t* __restrict__ outb,
    int M, int N, int K, int act) {
  int lane = threadIdx.x & 31;
  int wave = threadIdx.x >> 5;
  int tiles_n = N >> 5;
  int t  = blockIdx.x * 8 + wave;
  if (t >= (M >> 5) * tiles_n) return;
  int tm = t / tiles_n, tn = t - tm * tiles_n;
  int row0 = tm << 5, col0 = tn << 5;

  v8f a00 = {}, a01 = {}, a10 = {}, a11 = {};
  for (int k0 = 0; k0 < K; k0 += 32) {
    v16bf fa0 = load_a_frag(A, K, row0,      k0, lane);
    v16bf fa1 = load_a_frag(A, K, row0 + 16, k0, lane);
    v16bf fb0 = load_b_frag(W, K, col0,      k0, lane);
    v16bf fb1 = load_b_frag(W, K, col0 + 16, k0, lane);
    a00 = WMMA_BF16(fa0, fb0, a00);
    a01 = WMMA_BF16(fa0, fb1, a01);
    a10 = WMMA_BF16(fa1, fb0, a10);
    a11 = WMMA_BF16(fa1, fb1, a11);
  }
  store_tile(a00, row0,      col0,      N, bias, addsrc, outf, outb, act, lane);
  store_tile(a01, row0,      col0 + 16, N, bias, addsrc, outf, outb, act, lane);
  store_tile(a10, row0 + 16, col0,      N, bias, addsrc, outf, outb, act, lane);
  store_tile(a11, row0 + 16, col0 + 16, N, bias, addsrc, outf, outb, act, lane);
}

// ---------------------------------------------------------------------------
// Fused attention-logit + top-8 + softmax: never materializes the 8192x8192
// logit matrix. Wave owns 16 rows; e_h fragments stay in VGPRs; e_t column
// tiles streamed from L2. Each wave uses a private LDS slice, and per-wave
// LDS ops complete in order (ISA 7.3), so only a compiler scheduling fence
// (wave_barrier) is needed -- no block barriers in the 512-iteration loop.
// SCALE is applied only inside the final softmax (monotone w.r.t. top-k).
// ---------------------------------------------------------------------------
__global__ __launch_bounds__(256) void attn_topk_kernel(
    const bf16_t* __restrict__ ehb, const bf16_t* __restrict__ etb,
    int* __restrict__ tki, float* __restrict__ tkp) {
  __shared__ float sval[8][256];
  __shared__ int   sidx[8][256];
  int lane = threadIdx.x & 31;
  int wave = threadIdx.x >> 5;
  int tm   = blockIdx.x * 8 + wave;
  int row0 = tm << 4;

  v16bf afrag[8];
#pragma unroll
  for (int kt = 0; kt < 8; ++kt)
    afrag[kt] = load_a_frag(ehb, DDIM, row0, kt * 32, lane);

  float tv[8]; int ti[8];
#pragma unroll
  for (int j = 0; j < 8; ++j) { tv[j] = -3.0e38f; ti[j] = 0; }
  float curmin = -3.0e38f;

  int r     = lane & 15;          // row this lane maintains
  int chalf = (lane >> 4) << 3;   // which 8 columns of the tile it scans
  int col   = lane & 15;
  int rb    = (lane >> 4) << 3;

  for (int ct = 0; ct < NPATCH / 16; ++ct) {
    v8f acc = {};
#pragma unroll
    for (int kt = 0; kt < 8; ++kt) {
      v16bf b = load_b_frag(etb, DDIM, ct << 4, kt * 32, lane);
      acc = WMMA_BF16(afrag[kt], b, acc);
    }
#pragma unroll
    for (int j = 0; j < 8; ++j) sval[wave][(rb + j) * 16 + col] = acc[j];
    __builtin_amdgcn_wave_barrier();   // intra-wave LDS store->load ordering
#pragma unroll
    for (int c = 0; c < 8; ++c) {
      float v = sval[wave][r * 16 + chalf + c];
      if (v > curmin) {                    // early-out: rare path
        int mslot = 0; float mv = tv[0];
#pragma unroll
        for (int j = 1; j < 8; ++j) if (tv[j] < mv) { mv = tv[j]; mslot = j; }
        tv[mslot] = v; ti[mslot] = (ct << 4) + chalf + c;
        mv = tv[0];
#pragma unroll
        for (int j = 1; j < 8; ++j) mv = fminf(mv, tv[j]);
        curmin = mv;
      }
    }
    __builtin_amdgcn_wave_barrier();
  }

  // merge the two half-lane candidate lists, select top-8, softmax(SCALE*w)
#pragma unroll
  for (int j = 0; j < 8; ++j) {
    sval[wave][r * 16 + chalf + j] = tv[j];
    sidx[wave][r * 16 + chalf + j] = ti[j];
  }
  __builtin_amdgcn_wave_barrier();
  if (lane < 16) {
    float vals[16]; int idxs[16];
#pragma unroll
    for (int j = 0; j < 16; ++j) {
      vals[j] = sval[wave][r * 16 + j];
      idxs[j] = sidx[wave][r * 16 + j];
    }
    float w[8]; int wi[8];
#pragma unroll
    for (int s = 0; s < 8; ++s) {
      int best = 0; float bv = vals[0];
#pragma unroll
      for (int j = 1; j < 16; ++j) if (vals[j] > bv) { bv = vals[j]; best = j; }
      w[s] = bv; wi[s] = idxs[best]; vals[best] = -3.0e38f;
    }
    float Z = 0.0f, p[8];
#pragma unroll
    for (int s = 0; s < 8; ++s) { p[s] = __expf(SCALEF * (w[s] - w[0])); Z += p[s]; }
    float inv = 1.0f / Z;
    int row = row0 + r;
#pragma unroll
    for (int s = 0; s < 8; ++s) {
      tkp[row * KTOP + s] = p[s] * inv;
      tki[row * KTOP + s] = wi[s];
    }
  }
}

// ---------------------------------------------------------------------------
// Gather + gate: Nb_h = e_t[idx]; gate = tanh(e_h + p*Nb + (1-p)*e_h);
// ka softmax; e_Nh; emit bf16 (e_h+e_Nh) and (e_h*e_Nh) for l1/l2 GEMMs.
// One block per patch, thread = feature dim. Wave shfl reductions + a single
// cross-wave LDS combine (2 block barriers total instead of ~70).
// ---------------------------------------------------------------------------
__global__ __launch_bounds__(256) void gate_kernel(
    const float* __restrict__ eh, const float* __restrict__ et,
    const int* __restrict__ tki, const float* __restrict__ tkp,
    bf16_t* __restrict__ sum_in, bf16_t* __restrict__ bi_in) {
  __shared__ float part[KTOP * 8];   // [k][wave]
  __shared__ float kp[KTOP];
  int n = blockIdx.x, d = threadIdx.x;
  int lane = d & 31, wave = d >> 5;
  float ehv = eh[(size_t)n * DDIM + d];
  float nb[KTOP];
#pragma unroll
  for (int k = 0; k < KTOP; ++k) {
    int idx   = tki[n * KTOP + k];
    float nbv = et[(size_t)idx * DDIM + d];
    float p   = tkp[n * KTOP + k];
    float ehr = p * nbv + (1.0f - p) * ehv;
    float g   = tanhf(ehv + ehr);
    nb[k] = nbv;
    float v = nbv * g;
#pragma unroll
    for (int off = 16; off > 0; off >>= 1) v += __shfl_xor(v, off, 32);
    if (lane == 0) part[k * 8 + wave] = v;
  }
  __syncthreads();
  if (d == 0) {
    float ka[KTOP];
#pragma unroll
    for (int k = 0; k < KTOP; ++k) {
      float s = 0.0f;
#pragma unroll
      for (int w = 0; w < 8; ++w) s += part[k * 8 + w];
      ka[k] = s;
    }
    float m = ka[0];
#pragma unroll
    for (int k = 1; k < KTOP; ++k) m = fmaxf(m, ka[k]);
    float Z = 0.0f, e[KTOP];
#pragma unroll
    for (int k = 0; k < KTOP; ++k) { e[k] = __expf(ka[k] - m); Z += e[k]; }
#pragma unroll
    for (int k = 0; k < KTOP; ++k) kp[k] = e[k] / Z;
  }
  __syncthreads();
  float eNh = 0.0f;
#pragma unroll
  for (int k = 0; k < KTOP; ++k) eNh += kp[k] * nb[k];
  size_t o = (size_t)n * DDIM + d;
  sum_in[o] = (bf16_t)(ehv + eNh);
  bi_in[o]  = (bf16_t)(ehv * eNh);
}

// s[n] = dot(hid[n,:128], att2_w) + att2_b  (one wave per row)
__global__ __launch_bounds__(256) void att_score_kernel(
    const float* __restrict__ hid, const float* __restrict__ w2,
    const float* __restrict__ b2, float* __restrict__ s) {
  int lane = threadIdx.x & 31;
  int n    = blockIdx.x * 8 + (threadIdx.x >> 5);
  float acc = 0.0f;
  for (int i = lane; i < 128; i += 32) acc += hid[(size_t)n * 128 + i] * w2[i];
  for (int off = 16; off > 0; off >>= 1) acc += __shfl_down(acc, off, 32);
  if (lane == 0) s[n] = acc + b2[0];
}

// global softmax statistics over 8192 scores (single block)
__global__ __launch_bounds__(1024) void softmax_stats_kernel(
    const float* __restrict__ s, float* __restrict__ red) {
  __shared__ float sm[1024];
  int t = threadIdx.x;
  float m = -3.0e38f;
  for (int i = t; i < NPATCH; i += 1024) m = fmaxf(m, s[i]);
  sm[t] = m; __syncthreads();
  for (int st = 512; st > 0; st >>= 1) {
    if (t < st) sm[t] = fmaxf(sm[t], sm[t + st]);
    __syncthreads();
  }
  float mx = sm[0]; __syncthreads();
  float z = 0.0f;
  for (int i = t; i < NPATCH; i += 1024) z += __expf(s[i] - mx);
  sm[t] = z; __syncthreads();
  for (int st = 512; st > 0; st >>= 1) {
    if (t < st) sm[t] += sm[t + st];
    __syncthreads();
  }
  if (t == 0) { red[0] = mx; red[1] = sm[0]; }
}

// cls[d] = sum_n softmax(s)[n] * e_h_out[n,d]  (one block per feature dim)
__global__ __launch_bounds__(256) void cls_kernel(
    const float* __restrict__ ehout, const float* __restrict__ s,
    const float* __restrict__ red, float* __restrict__ cls) {
  __shared__ float sm[256];
  int d = blockIdx.x, t = threadIdx.x;
  float mx = red[0], invZ = 1.0f / red[1];
  float acc = 0.0f;
  for (int n = t; n < NPATCH; n += 256)
    acc += __expf(s[n] - mx) * invZ * ehout[(size_t)n * DDIM + d];
  sm[t] = acc; __syncthreads();
  for (int st = 128; st > 0; st >>= 1) {
    if (t < st) sm[t] += sm[t + st];
    __syncthreads();
  }
  if (t == 0) cls[d] = sm[0];
}

__global__ void cvt_kernel(const float* __restrict__ in,
                           bf16_t* __restrict__ out, int n) {
  int i = blockIdx.x * blockDim.x + threadIdx.x;
  if (i < n) out[i] = (bf16_t)in[i];
}

// ---------------------------------------------------------------------------
// workspace layout (all offsets 256B aligned), total ~77.3 MB
// ---------------------------------------------------------------------------
#define OFF_XB   ((size_t)0)          // x_path bf16        16 MB
#define OFF_H1B  ((size_t)16777216)   // h1 bf16             4 MB
#define OFF_H2B  ((size_t)20971520)   // h2 bf16             4 MB
#define OFF_EHF  ((size_t)25165824)   // e_h f32             8 MB
#define OFF_ETF  ((size_t)33554432)   // e_t f32             8 MB
#define OFF_EHB  ((size_t)41943040)   // e_h bf16            4 MB
#define OFF_ETB  ((size_t)46137344)   // e_t bf16            4 MB
#define OFF_W1B  ((size_t)50331648)   // fc1_w bf16        512 KB
#define OFF_W2B  ((size_t)50855936)   // fc2_w bf16        128 KB
#define OFF_WHB  ((size_t)50987008)
#define OFF_WTB  ((size_t)51118080)
#define OFF_L1B  ((size_t)51249152)
#define OFF_L2B  ((size_t)51380224)
#define OFF_A1B  ((size_t)51511296)   // att1_w bf16        64 KB
#define OFF_TKI  ((size_t)51576832)   // topk idx          256 KB
#define OFF_TKP  ((size_t)51838976)   // topk prob         256 KB
#define OFF_SUMB ((size_t)52101120)   // (e_h+e_Nh) bf16     4 MB
#define OFF_BIB  ((size_t)56295424)   // (e_h*e_Nh) bf16     4 MB
#define OFF_SEMF ((size_t)60489728)   // relu(sum_emb) f32   8 MB
#define OFF_EOB  ((size_t)68878336)   // e_h_out bf16        4 MB
#define OFF_HIDF ((size_t)73072640)   // att hidden f32      4 MB
#define OFF_S    ((size_t)77266944)   // scores             32 KB
#define OFF_RED  ((size_t)77299712)   // {max, Z}

extern "C" void kernel_launch(void* const* d_in, const int* in_sizes, int n_in,
                              void* d_out, int out_size, void* d_ws, size_t ws_size,
                              hipStream_t stream) {
  const float* x_path = (const float*)d_in[0];
  const float* fc1_w = (const float*)d_in[1];  const float* fc1_b = (const float*)d_in[2];
  const float* fc2_w = (const float*)d_in[3];  const float* fc2_b = (const float*)d_in[4];
  const float* wh_w  = (const float*)d_in[5];  const float* wh_b  = (const float*)d_in[6];
  const float* wt_w  = (const float*)d_in[7];  const float* wt_b  = (const float*)d_in[8];
  const float* l1_w  = (const float*)d_in[9];  const float* l1_b  = (const float*)d_in[10];
  const float* l2_w  = (const float*)d_in[11]; const float* l2_b  = (const float*)d_in[12];
  const float* att1_w = (const float*)d_in[13]; const float* att1_b = (const float*)d_in[14];
  const float* att2_w = (const float*)d_in[15]; const float* att2_b = (const float*)d_in[16];

  char* ws = (char*)d_ws;
  bf16_t* xb   = (bf16_t*)(ws + OFF_XB);
  bf16_t* h1b  = (bf16_t*)(ws + OFF_H1B);
  bf16_t* h2b  = (bf16_t*)(ws + OFF_H2B);
  float*  ehf  = (float*)(ws + OFF_EHF);
  float*  etf  = (float*)(ws + OFF_ETF);
  bf16_t* ehb  = (bf16_t*)(ws + OFF_EHB);
  bf16_t* etb  = (bf16_t*)(ws + OFF_ETB);
  bf16_t* w1b  = (bf16_t*)(ws + OFF_W1B);
  bf16_t* w2b  = (bf16_t*)(ws + OFF_W2B);
  bf16_t* whb  = (bf16_t*)(ws + OFF_WHB);
  bf16_t* wtb  = (bf16_t*)(ws + OFF_WTB);
  bf16_t* l1b  = (bf16_t*)(ws + OFF_L1B);
  bf16_t* l2b  = (bf16_t*)(ws + OFF_L2B);
  bf16_t* a1b  = (bf16_t*)(ws + OFF_A1B);
  int*    tki  = (int*)(ws + OFF_TKI);
  float*  tkp  = (float*)(ws + OFF_TKP);
  bf16_t* sumb = (bf16_t*)(ws + OFF_SUMB);
  bf16_t* bib  = (bf16_t*)(ws + OFF_BIB);
  float*  semf = (float*)(ws + OFF_SEMF);
  bf16_t* eob  = (bf16_t*)(ws + OFF_EOB);
  float*  hidf = (float*)(ws + OFF_HIDF);
  float*  sN   = (float*)(ws + OFF_S);
  float*  red  = (float*)(ws + OFF_RED);

  float* ehout = (float*)d_out;                    // [8192, 256]
  float* cls   = ehout + (size_t)NPATCH * DDIM;    // [256]

  auto cvt = [&](const float* src, bf16_t* dst, int n) {
    cvt_kernel<<<(n + 255) / 256, 256, 0, stream>>>(src, dst, n);
  };
  cvt(x_path, xb, NPATCH * 1024);
  cvt(fc1_w, w1b, DDIM * 1024);
  cvt(fc2_w, w2b, DDIM * DDIM);
  cvt(wh_w,  whb, DDIM * DDIM);
  cvt(wt_w,  wtb, DDIM * DDIM);
  cvt(l1_w,  l1b, DDIM * DDIM);
  cvt(l2_w,  l2b, DDIM * DDIM);
  cvt(att1_w, a1b, 128 * DDIM);

  auto gemm = [&](const bf16_t* A, const bf16_t* W, const float* bias,
                  const float* addsrc, float* outf, bf16_t* outb,
                  int M, int N, int K, int act) {
    int tiles = (M >> 5) * (N >> 5);   // 32x32 tile per wave
    gemm_bf16_kernel<<<(tiles + 7) / 8, 256, 0, stream>>>(
        A, W, bias, addsrc, outf, outb, M, N, K, act);
  };

  // pathology_fc
  gemm(xb,  w1b, fc1_b, nullptr, nullptr, h1b, NPATCH, DDIM, 1024, 1);
  gemm(h1b, w2b, fc2_b, nullptr, nullptr, h2b, NPATCH, DDIM, DDIM, 1);
  // head / tail embeddings (fp32 + bf16 copies)
  gemm(h2b, whb, wh_b, nullptr, ehf, ehb, NPATCH, DDIM, DDIM, 0);
  gemm(h2b, wtb, wt_b, nullptr, etf, etb, NPATCH, DDIM, DDIM, 0);

  // fused N x N similarity + top-8 + neighbor softmax (no logit matrix)
  attn_topk_kernel<<<NPATCH / 16 / 8, 256, 0, stream>>>(ehb, etb, tki, tkp);

  // neighbor gather + gating + ka softmax + e_Nh; emits l1/l2 GEMM inputs
  gate_kernel<<<NPATCH, 256, 0, stream>>>(ehf, etf, tki, tkp, sumb, bib);

  // e_h_out = relu(sum@l1^T+b) + relu(bi@l2^T+b)
  gemm(sumb, l1b, l1_b, nullptr, semf, nullptr, NPATCH, DDIM, DDIM, 1);
  gemm(bib,  l2b, l2_b, semf, ehout, eob, NPATCH, DDIM, DDIM, 1);

  // GlobalAttention readout
  gemm(eob, a1b, att1_b, nullptr, hidf, nullptr, NPATCH, 128, DDIM, 2);
  att_score_kernel<<<NPATCH / 8, 256, 0, stream>>>(hidf, att2_w, att2_b, sN);
  softmax_stats_kernel<<<1, 1024, 0, stream>>>(sN, red);
  cls_kernel<<<DDIM, 256, 0, stream>>>(ehout, sN, red, cls);
}
